// CayleyConvED_68856915689932
// MI455X (gfx1250) — matrix operand order repaired and checked by reading, not compiled, not run on hardware
//
#include <hip/hip_runtime.h>

// CayleyConvED for MI455X (gfx1250, wave32).
// rfft2(x) -> M=I+A (f16) -> R=2X-MX -> 5x Richardson Z=R+Z-MZ -> Yf=H*Z -> irfft2+bias
// Complex GEMMs: v_wmma_f32_16x16x32_f16; RHS staged into LDS by the Tensor Data Mover
// (tensor_load_to_lds + s_wait_tensorcnt) with TDM hardware padding for bank-conflict-free
// fragment reads (ds_load_b128).

typedef _Float16     v16h __attribute__((ext_vector_type(16)));
typedef _Float16     h8   __attribute__((ext_vector_type(8)));
typedef float        v8f  __attribute__((ext_vector_type(8)));
typedef unsigned int u32x4 __attribute__((ext_vector_type(4)));
typedef int          i32x8 __attribute__((ext_vector_type(8)));
typedef int          i32x4 __attribute__((ext_vector_type(4)));

#define NN    32
#define NF    17
#define FREQ  544           // 32*17
#define CIN   256
#define BB    16
#define VECF  (FREQ * CIN * BB)
#define LROW  264           // LDS row stride in halves: 256 data + 8 pad (528 B)

__device__ __forceinline__ void fill_trig(float* ct, float* st) {
    int t = threadIdx.x;
    if (t < 32) {
        float ang = (2.0f * 3.14159265358979323846f / 32.0f) * (float)t;
        ct[t] = __cosf(ang);
        st[t] = __sinf(ang);
    }
    __syncthreads();
}

// ---- TDM descriptor builders (CDNA5 ISA ch.8: D# group0/group1) ----
__device__ __forceinline__ u32x4 tdm_g0(unsigned int lds_addr, const void* gaddr) {
    unsigned long long ga = (unsigned long long)(size_t)gaddr;
    u32x4 g;
    g.x = 1u;                                                // count=1 (valid user D#)
    g.y = lds_addr;                                          // lds_addr[31:0]
    g.z = (unsigned int)ga;                                  // global_addr[31:0]
    g.w = (unsigned int)((ga >> 32) & 0x01FFFFFFu) | (2u << 30); // addr[56:32] | type=2
    return g;
}
__device__ __forceinline__ i32x8 tdm_g1_1d_pad(unsigned int nbytes) {
    // 1-D tile of `nbytes` bytes (data_size=1B); LDS pad: 16B after every 512B.
    i32x8 g = {};
    g[0] = (0 << 16)      // data_size = 1 byte
         | (1 << 20)      // pad_enable
         | (6 << 22)      // pad_interval: 128 DWORDs = 512 B
         | (3 << 25);     // pad_amount:   4 DWORDs  = 16 B
    g[1] = (int)((nbytes & 0xFFFFu) << 16);        // tensor_dim0[15:0]  -> bits 63:48
    g[2] = (int)((nbytes >> 16) & 0xFFFFu)         // tensor_dim0[31:16] -> bits 79:64
         | (1 << 16);                              // tensor_dim1 = 1    -> bits 95:80
    g[3] = (int)((nbytes & 0xFFFFu) << 16);        // tile_dim0          -> bits 127:112
    g[4] = 0;                                      // tile_dim1/2 unused
    g[5] = (int)nbytes;                            // tensor_dim0_stride lo32
    g[6] = 0;
    g[7] = 0;
    return g;
}

// ---------------- forward rfft2 of x -> transposed f16 [f][b][cin] ----------------
__global__ __launch_bounds__(256)
void k_rfft2_x(const float* __restrict__ x,
               _Float16* __restrict__ Xtr, _Float16* __restrict__ Xti) {
    __shared__ float ct[32], st[32];
    __shared__ float img[1024];
    __shared__ float tre[FREQ], tim[FREQ];
    fill_trig(ct, st);
    const int blk = blockIdx.x;            // b*256 + c
    const int c = blk & 255, b = blk >> 8;
    const float* src = x + (size_t)blk * 1024;
    for (int i = threadIdx.x; i < 1024; i += 256) img[i] = src[i];
    __syncthreads();
    for (int i = threadIdx.x; i < FREQ; i += 256) {
        int r = i / NF, v = i - r * NF;
        float sr = 0.f, si = 0.f;
        for (int q = 0; q < 32; ++q) {
            int k = (v * q) & 31;
            float val = img[r * 32 + q];
            sr += val * ct[k];
            si -= val * st[k];
        }
        tre[i] = sr; tim[i] = si;
    }
    __syncthreads();
    for (int i = threadIdx.x; i < FREQ; i += 256) {
        int u = i / NF, v = i - u * NF;
        float sr = 0.f, si = 0.f;
        for (int r = 0; r < 32; ++r) {
            int k = (u * r) & 31;
            float tr = tre[r * NF + v], ti = tim[r * NF + v];
            sr += tr * ct[k] + ti * st[k];
            si += ti * ct[k] - tr * st[k];
        }
        size_t o = ((size_t)i * BB + b) * CIN + c;   // [f][b][cin]
        Xtr[o] = (_Float16)sr;
        Xti[o] = (_Float16)si;
    }
}

// ---------------- ||wfft||^2 partial sums ----------------
__global__ __launch_bounds__(256)
void k_wnorm(const float* __restrict__ w, float* __restrict__ partial) {
    __shared__ float ct[32], st[32];
    __shared__ float red[256];
    fill_trig(ct, st);
    const int ij = blockIdx.x * 256 + threadIdx.x;
    const float* wp = w + (size_t)ij * 9;
    float wv[9];
#pragma unroll
    for (int t = 0; t < 9; ++t) wv[t] = wp[t];
    float acc = 0.f;
    for (int f = 0; f < FREQ; ++f) {
        int u = f / NF, v = f - u * NF;
        float re = 0.f, im = 0.f;
#pragma unroll
        for (int p = 0; p < 3; ++p)
#pragma unroll
            for (int q = 0; q < 3; ++q) {
                int k = (u * p + v * q) & 31;
                float val = wv[p * 3 + q];
                re += val * ct[k];
                im -= val * st[k];
            }
        acc += re * re + im * im;
    }
    red[threadIdx.x] = acc;
    __syncthreads();
    for (int s = 128; s > 0; s >>= 1) {
        if (threadIdx.x < (unsigned)s) red[threadIdx.x] += red[threadIdx.x + s];
        __syncthreads();
    }
    if (threadIdx.x == 0) partial[blockIdx.x] = red[0];
}

// ---------------- build M = I + A in f16 (one block per frequency) ----------------
__global__ __launch_bounds__(256)
void k_buildM(const float* __restrict__ w, const float* __restrict__ alpha,
              const float* __restrict__ partial,
              _Float16* __restrict__ Mr, _Float16* __restrict__ Mi) {
    __shared__ float ct[32], st[32];
    __shared__ float red[256];
    fill_trig(ct, st);
    red[threadIdx.x] = partial[threadIdx.x];
    __syncthreads();
    for (int s = 128; s > 0; s >>= 1) {
        if (threadIdx.x < (unsigned)s) red[threadIdx.x] += red[threadIdx.x + s];
        __syncthreads();
    }
    const float n2 = red[0] > 0.f ? red[0] : 1.f;
    const float sscale = alpha[0] * rsqrtf(n2);
    const int f = blockIdx.x;
    const int u = f / NF, v = f - u * NF;
    const int ksh = (u + v) & 31;
    const float shr = ct[ksh], shi = -st[ksh];
    _Float16* mrp = Mr + (size_t)f * (CIN * CIN);
    _Float16* mip = Mi + (size_t)f * (CIN * CIN);
    for (int idx = threadIdx.x; idx < CIN * CIN; idx += 256) {
        int i = idx >> 8, j = idx & 255;
        const float* wij = w + ((size_t)i * CIN + j) * 9;
        const float* wji = w + ((size_t)j * CIN + i) * 9;
        float re1 = 0.f, im1 = 0.f, re2 = 0.f, im2 = 0.f;
#pragma unroll
        for (int p = 0; p < 3; ++p)
#pragma unroll
            for (int q = 0; q < 3; ++q) {
                int k = (u * p + v * q) & 31;
                float cs = ct[k], sn = st[k];
                float a  = wij[p * 3 + q];
                float bq = wji[p * 3 + q];
                re1 += a * cs;  im1 -= a * sn;
                re2 += bq * cs; im2 -= bq * sn;
            }
        float w1r = shr * re1 + shi * im1, w1i = shi * re1 - shr * im1;
        float w2r = shr * re2 + shi * im2, w2i = shi * re2 - shr * im2;
        float ar = sscale * (w1r - w2r);
        float ai = sscale * (w1i + w2i);
        float mr = (i == j ? 1.0f : 0.0f) + ar;
        mrp[idx] = (_Float16)mr;
        mip[idx] = (_Float16)ai;
    }
}

// ---------------- batched complex GEMM ----------------
// RICH=1: A = M (f16 planar [f][256][256]); OUT(f16,T) = P + Q - A*B   (Richardson)
// RICH=0: A = H (f32 planar);               OUT(f32 planar) = A*B
// B/P/Q: transposed f16 [f][16][256]. One wave = one 16(M)x16(N) tile, K=256.
template <bool RICH>
__global__ __launch_bounds__(256)
void k_cgemm(const _Float16* __restrict__ A16r, const _Float16* __restrict__ A16i,
             const float* __restrict__ A32r, const float* __restrict__ A32i,
             const _Float16* __restrict__ Btr, const _Float16* __restrict__ Bti,
             const _Float16* __restrict__ Ptr, const _Float16* __restrict__ Pti,
             const _Float16* __restrict__ Qtr, const _Float16* __restrict__ Qti,
             _Float16* __restrict__ Ctr, _Float16* __restrict__ Cti,
             float* __restrict__ Cr, float* __restrict__ Ci) {
    __shared__ __align__(64) _Float16 lbr[BB * LROW];   // padded: row stride 528 B
    __shared__ __align__(64) _Float16 lbi[BB * LROW];
    const int f    = blockIdx.x >> 1;
    const int half = blockIdx.x & 1;
    const int tid  = threadIdx.x;
    const size_t bofs16 = (size_t)f * (BB * CIN);       // halves into [f][16][256]

    // ---- stage RHS tile (2 x 8 KB) into LDS via the Tensor Data Mover ----
#if __has_builtin(__builtin_amdgcn_tensor_load_to_lds) && __has_builtin(__builtin_amdgcn_s_wait_tensorcnt)
    if (tid < 32) {   // one DMA pair per workgroup (TDM ignores EXEC; issue from wave 0 only)
        const i32x8 g1 = tdm_g1_1d_pad(BB * CIN * 2u);  // 8192 B, pad 16B/512B
        const i32x4 z4 = {};
        const i32x8 z8 = {};
        u32x4 g0r = tdm_g0((unsigned int)(size_t)&lbr[0], (const void*)(Btr + bofs16));
        u32x4 g0i = tdm_g0((unsigned int)(size_t)&lbi[0], (const void*)(Bti + bofs16));
        __builtin_amdgcn_tensor_load_to_lds(g0r, g1, z4, z4, z8, 0);
        __builtin_amdgcn_tensor_load_to_lds(g0i, g1, z4, z4, z8, 0);
        __builtin_amdgcn_s_wait_tensorcnt(0);
    }
#else
    for (int g = tid; g < BB * CIN / 8; g += 256) {     // 16B chunks
        int row = g >> 5, c = g & 31;
        *(h8*)(lbr + row * LROW + c * 8) = ((const h8*)(Btr + bofs16))[g];
        *(h8*)(lbi + row * LROW + c * 8) = ((const h8*)(Bti + bofs16))[g];
    }
#endif
    __syncthreads();

    const int wave    = tid >> 5;
    const int lane    = tid & 31;
    const int mtile   = half * 8 + wave;
    const int rowA    = mtile * 16 + (lane & 15);
    const int colbase = (lane >> 4) << 3;   // A frag: K 0/8 (+16 for upper 8 elems)
    const int khalfB  = (lane >> 4) << 3;   // B frag: K 0/8 (+16 for upper 8 elems)
    const int nB      = lane & 15;

    v8f accRR = {}; v8f accII = {}; v8f accRI = {}; v8f accIR = {};
    const size_t aofs = ((size_t)f * CIN + rowA) * CIN;

    for (int k0 = 0; k0 < CIN; k0 += 32) {
        v16h ar, ai;
        if (RICH) {
            const h8* pr = (const h8*)(A16r + aofs + k0 + colbase);
            const h8* pi = (const h8*)(A16i + aofs + k0 + colbase);
            ((h8*)&ar)[0] = pr[0];
            ((h8*)&ar)[1] = pr[2];
            ((h8*)&ai)[0] = pi[0];
            ((h8*)&ai)[1] = pi[2];
            if (k0 + 32 < CIN) {
                __builtin_prefetch(A16r + aofs + k0 + 32, 0, 0);
                __builtin_prefetch(A16i + aofs + k0 + 32, 0, 0);
            }
        } else {
            const float* pr = A32r + aofs + k0 + colbase;
            const float* pi = A32i + aofs + k0 + colbase;
#pragma unroll
            for (int e = 0; e < 8; ++e) {
                ar[e]     = (_Float16)pr[e];
                ar[e + 8] = (_Float16)pr[e + 16];
                ai[e]     = (_Float16)pi[e];
                ai[e + 8] = (_Float16)pi[e + 16];
            }
            if (k0 + 32 < CIN) {
                __builtin_prefetch(A32r + aofs + k0 + 32, 0, 0);
                __builtin_prefetch(A32i + aofs + k0 + 32, 0, 0);
            }
        }
        // B fragment: K 0/8 and +16 (8 halves each) -> two aligned 16B LDS loads
        v16h br_, bi_;
        {
            const h8* lr = (const h8*)(lbr + nB * LROW + k0 + khalfB);
            const h8* li = (const h8*)(lbi + nB * LROW + k0 + khalfB);
            ((h8*)&br_)[0] = lr[0];
            ((h8*)&br_)[1] = lr[2];
            ((h8*)&bi_)[0] = li[0];
            ((h8*)&bi_)[1] = li[2];
        }
        accRR = __builtin_amdgcn_wmma_f32_16x16x32_f16(false, ar, false, br_, (short)0, accRR, false, false);
        accII = __builtin_amdgcn_wmma_f32_16x16x32_f16(false, ai, false, bi_, (short)0, accII, false, false);
        accRI = __builtin_amdgcn_wmma_f32_16x16x32_f16(false, ar, false, bi_, (short)0, accRI, false, false);
        accIR = __builtin_amdgcn_wmma_f32_16x16x32_f16(false, ai, false, br_, (short)0, accIR, false, false);
    }

    const int mrow0 = mtile * 16 + ((lane >> 4) << 3);
    if (RICH) {
        const size_t tofs = bofs16 + (size_t)nB * CIN + mrow0;  // transposed [f][n][m]
        h8 pr = *(const h8*)(Ptr + tofs);
        h8 pi = *(const h8*)(Pti + tofs);
        h8 qr = *(const h8*)(Qtr + tofs);
        h8 qi = *(const h8*)(Qti + tofs);
        h8 outr, outi;
#pragma unroll
        for (int r = 0; r < 8; ++r) {
            float gr = (float)pr[r] + (float)qr[r] - (accRR[r] - accII[r]);
            float gi = (float)pi[r] + (float)qi[r] - (accRI[r] + accIR[r]);
            outr[r] = (_Float16)gr;
            outi[r] = (_Float16)gi;
        }
        *(h8*)(Ctr + tofs) = outr;
        *(h8*)(Cti + tofs) = outi;
    } else {
        const size_t pofs = (size_t)f * (CIN * BB);             // planar f32 [f][m][n]
#pragma unroll
        for (int r = 0; r < 8; ++r) {
            size_t idx = pofs + (size_t)(mrow0 + r) * BB + nB;
            Cr[idx] = accRR[r] - accII[r];
            Ci[idx] = accRI[r] + accIR[r];
        }
    }
}

// ---------------- irfft2 + bias ----------------
__global__ __launch_bounds__(256)
void k_irfft2(const float* __restrict__ Yr, const float* __restrict__ Yi,
              const float* __restrict__ bias, float* __restrict__ y) {
    __shared__ float ct[32], st[32];
    __shared__ float Sre[1024], Sim[1024];
    __shared__ float Tre[1024], Tim[1024];
    fill_trig(ct, st);
    const int blk = blockIdx.x;            // b*256 + co
    const int co = blk & 255, b = blk >> 8;
    for (int i = threadIdx.x; i < FREQ; i += 256) {
        int u = i / NF, v = i - u * NF;
        size_t o = (((size_t)i * CIN) + co) * BB + b;
        float re = Yr[o], im = Yi[o];
        Sre[u * 32 + v] = re; Sim[u * 32 + v] = im;
        if (v >= 1 && v <= 15) {
            int uu = (32 - u) & 31, vv = 32 - v;
            Sre[uu * 32 + vv] = re; Sim[uu * 32 + vv] = -im;
        }
    }
    __syncthreads();
    for (int i = threadIdx.x; i < 1024; i += 256) {
        int u = i >> 5, cx = i & 31;
        float sr = 0.f, si = 0.f;
        for (int v = 0; v < 32; ++v) {
            int k = (v * cx) & 31;
            float cs = ct[k], sn = st[k];
            float re = Sre[u * 32 + v], im = Sim[u * 32 + v];
            sr += re * cs - im * sn;
            si += re * sn + im * cs;
        }
        Tre[i] = sr; Tim[i] = si;
    }
    __syncthreads();
    const float bval = bias[co];
    float* dst = y + (size_t)blk * 1024;
    for (int i = threadIdx.x; i < 1024; i += 256) {
        int r = i >> 5, cx = i & 31;
        float sr = 0.f;
        for (int u = 0; u < 32; ++u) {
            int k = (u * r) & 31;
            sr += Tre[u * 32 + cx] * ct[k] - Tim[u * 32 + cx] * st[k];
        }
        dst[i] = sr * (1.0f / 1024.0f) + bval;
    }
}

extern "C" void kernel_launch(void* const* d_in, const int* in_sizes, int n_in,
                              void* d_out, int out_size, void* d_ws, size_t ws_size,
                              hipStream_t stream) {
    (void)in_sizes; (void)n_in; (void)out_size; (void)ws_size;
    const float* x     = (const float*)d_in[0];
    const float* wgt   = (const float*)d_in[1];
    const float* alpha = (const float*)d_in[2];
    const float* bias  = (const float*)d_in[3];
    const float* Hre   = (const float*)d_in[4];
    const float* Him   = (const float*)d_in[5];
    float* y = (float*)d_out;

    char* p = (char*)d_ws;
    const size_t hvb = (size_t)VECF * sizeof(_Float16);  // transposed f16 RHS buffer
    const size_t fvb = (size_t)VECF * sizeof(float);     // planar f32 buffer
    _Float16* Xtr = (_Float16*)p; p += hvb;
    _Float16* Xti = (_Float16*)p; p += hvb;
    _Float16* Rtr = (_Float16*)p; p += hvb;
    _Float16* Rti = (_Float16*)p; p += hvb;
    _Float16* Z0r = (_Float16*)p; p += hvb;
    _Float16* Z0i = (_Float16*)p; p += hvb;
    _Float16* Z1r = (_Float16*)p; p += hvb;
    _Float16* Z1i = (_Float16*)p; p += hvb;
    float* Yfr = (float*)p; p += fvb;
    float* Yfi = (float*)p; p += fvb;
    const size_t mb = (size_t)FREQ * CIN * CIN * sizeof(_Float16);
    _Float16* Mr = (_Float16*)p; p += mb;
    _Float16* Mi = (_Float16*)p; p += mb;
    float* partial = (float*)p; p += 256 * sizeof(float);

    k_rfft2_x<<<BB * CIN, 256, 0, stream>>>(x, Xtr, Xti);
    k_wnorm<<<256, 256, 0, stream>>>(wgt, partial);
    k_buildM<<<FREQ, 256, 0, stream>>>(wgt, alpha, partial, Mr, Mi);
    // R = 2X - M X
    k_cgemm<true><<<FREQ * 2, 256, 0, stream>>>(Mr, Mi, nullptr, nullptr,
        Xtr, Xti, Xtr, Xti, Xtr, Xti, Rtr, Rti, nullptr, nullptr);
    // Z_{k+1} = R + Z_k - M Z_k, Z_0 = R
    k_cgemm<true><<<FREQ * 2, 256, 0, stream>>>(Mr, Mi, nullptr, nullptr,
        Rtr, Rti, Rtr, Rti, Rtr, Rti, Z0r, Z0i, nullptr, nullptr);
    k_cgemm<true><<<FREQ * 2, 256, 0, stream>>>(Mr, Mi, nullptr, nullptr,
        Z0r, Z0i, Rtr, Rti, Z0r, Z0i, Z1r, Z1i, nullptr, nullptr);
    k_cgemm<true><<<FREQ * 2, 256, 0, stream>>>(Mr, Mi, nullptr, nullptr,
        Z1r, Z1i, Rtr, Rti, Z1r, Z1i, Z0r, Z0i, nullptr, nullptr);
    k_cgemm<true><<<FREQ * 2, 256, 0, stream>>>(Mr, Mi, nullptr, nullptr,
        Z0r, Z0i, Rtr, Rti, Z0r, Z0i, Z1r, Z1i, nullptr, nullptr);
    k_cgemm<true><<<FREQ * 2, 256, 0, stream>>>(Mr, Mi, nullptr, nullptr,
        Z1r, Z1i, Rtr, Rti, Z1r, Z1i, Z0r, Z0i, nullptr, nullptr);
    // Yf = H Z
    k_cgemm<false><<<FREQ * 2, 256, 0, stream>>>(nullptr, nullptr, Hre, Him,
        Z0r, Z0i, nullptr, nullptr, nullptr, nullptr, nullptr, nullptr, Yfr, Yfi);
    k_irfft2<<<BB * CIN, 256, 0, stream>>>(Yfr, Yfi, bias, y);
}
